// Regressor_18433999634719
// MI455X (gfx1250) — compile-verified, matrix operand order checked by reading, not run on hardware
//
#include <hip/hip_runtime.h>
#include <hip/hip_bf16.h>
#include <stdint.h>

// ---------------------------------------------------------------------------
// Types for CDNA5 WMMA (wave32)
// ---------------------------------------------------------------------------
typedef _Float16 half_t;
typedef _Float16 v16h __attribute__((ext_vector_type(16)));
typedef _Float16 v8h  __attribute__((ext_vector_type(8)));
typedef float    v8f  __attribute__((ext_vector_type(8)));

// Problem constants (from reference)
#define NS   3200      // B*G*T conv samples
#define NN   100000    // graph nodes
#define NE   400000    // graph edges
#define HID  256
#define ENC  512

// ---------------------------------------------------------------------------
// Helpers
// ---------------------------------------------------------------------------
__device__ __forceinline__ unsigned f2ord(float f) {
    unsigned u = __float_as_uint(f);
    return (u & 0x80000000u) ? ~u : (u | 0x80000000u);
}
__device__ __forceinline__ float ord2f(unsigned u) {
    unsigned b = (u & 0x80000000u) ? (u ^ 0x80000000u) : ~u;
    return __uint_as_float(b);
}

// Load one 16x32 f16 fragment (A-row-major, or B^T row == B column) per the
// CDNA5 16-bit A-matrix 16x32 VGPR layout: lanes 0-15 rows, VGPR0-3 = K 0..7,
// VGPR4-7 = K 16..23; lanes 16-31 same rows, K 8..15 / 24..31.
__device__ __forceinline__ v16h load_frag_nt(const half_t* __restrict__ base,
                                             int ld, int row0, int k0, int lane) {
    int l  = lane & 15;
    int hi = lane >> 4;   // 0 or 1
    const half_t* p = base + (size_t)(row0 + l) * (size_t)ld + k0 + hi * 8;
    v8h a = *(const v8h*)(p);
    v8h b = *(const v8h*)(p + 16);
    v16h r;
#pragma unroll
    for (int i = 0; i < 8; ++i) { r[i] = a[i]; r[8 + i] = b[i]; }
    return r;
}

// ---------------------------------------------------------------------------
// Generic NT-GEMM with WMMA:  C[M,N] = A[M,Kp] * B[N,Kp]^T  (+bias,BN,ReLU)
// A, B are f16 row-major with K padded to a multiple of 32 (zero filled).
// One wave computes a 16x64 output tile (4 accumulators, A reused 4x);
// 8 waves per block cover 128 rows. N must be a multiple of 64 (true for all
// GEMMs in this model: 64,128,256,512,768). M tiles are grid-guarded.
// ---------------------------------------------------------------------------
__global__ __launch_bounds__(256) void k_gemm_nt(
    const half_t* __restrict__ A, const half_t* __restrict__ B,
    int M, int N, int Kp,
    const float* __restrict__ bias, const float* __restrict__ bnsc,
    const float* __restrict__ bnsh, int relu,
    float* __restrict__ outF, half_t* __restrict__ outH)
{
    int lane  = threadIdx.x & 31;
    int wave  = threadIdx.x >> 5;
    int mtile = blockIdx.y * 8 + wave;
    if (mtile * 16 >= M) return;                 // wave-uniform exit
    int m0 = mtile * 16;
    int n0 = blockIdx.x * 64;

    v8f acc0 = {}; v8f acc1 = {}; v8f acc2 = {}; v8f acc3 = {};
    for (int k0 = 0; k0 < Kp; k0 += 32) {
        v16h a  = load_frag_nt(A, Kp, m0,      k0, lane);
        v16h b0 = load_frag_nt(B, Kp, n0,      k0, lane);
        v16h b1 = load_frag_nt(B, Kp, n0 + 16, k0, lane);
        v16h b2 = load_frag_nt(B, Kp, n0 + 32, k0, lane);
        v16h b3 = load_frag_nt(B, Kp, n0 + 48, k0, lane);
        acc0 = __builtin_amdgcn_wmma_f32_16x16x32_f16(false, a, false, b0,
                                                      (short)0, acc0, false, false);
        acc1 = __builtin_amdgcn_wmma_f32_16x16x32_f16(false, a, false, b1,
                                                      (short)0, acc1, false, false);
        acc2 = __builtin_amdgcn_wmma_f32_16x16x32_f16(false, a, false, b2,
                                                      (short)0, acc2, false, false);
        acc3 = __builtin_amdgcn_wmma_f32_16x16x32_f16(false, a, false, b3,
                                                      (short)0, acc3, false, false);
    }

    // C/D layout: VGPR r -> row m0 + r (+8 for lanes 16-31); col = n0 + lane%16
    int colb  = n0 + (lane & 15);
    int rbase = m0 + ((lane >> 4) ? 8 : 0);
#pragma unroll
    for (int t = 0; t < 4; ++t) {
        v8f acc = (t == 0) ? acc0 : (t == 1) ? acc1 : (t == 2) ? acc2 : acc3;
        int col = colb + t * 16;
        float bi = bias ? bias[col] : 0.f;
        float sc = bnsc ? bnsc[col] : 1.f;
        float sh = bnsh ? bnsh[col] : 0.f;
#pragma unroll
        for (int r = 0; r < 8; ++r) {
            float v = acc[r] + bi;
            v = v * sc + sh;
            if (relu) v = v > 0.f ? v : 0.f;
            size_t idx = (size_t)(rbase + r) * (size_t)N + col;
            if (outF) outF[idx] = v;
            else      outH[idx] = (half_t)v;
        }
    }
}

// ---------------------------------------------------------------------------
// Utility kernels
// ---------------------------------------------------------------------------
__global__ __launch_bounds__(256) void k_fill_u32(unsigned* p, unsigned v, long n) {
    long i = (long)blockIdx.x * 256 + threadIdx.x;
    if (i < n) p[i] = v;
}

// f32 -> f16 with K padding (weights)
__global__ __launch_bounds__(256) void k_cvt_pad(const float* __restrict__ src,
                                                 half_t* __restrict__ dst,
                                                 int rows, int K, int Kp) {
    long i = (long)blockIdx.x * 256 + threadIdx.x;
    long tot = (long)rows * Kp;
    if (i >= tot) return;
    int c = (int)(i % Kp); long r = i / Kp;
    dst[i] = (c < K) ? (half_t)src[r * (long)K + c] : (half_t)0.f;
}

// BN fold: scale = g*rsqrt(v+eps), shift = b - m*scale
__global__ __launch_bounds__(256) void k_bnprep(const float* g, const float* b,
                                                const float* m, const float* v,
                                                float* sc, float* sh, int C) {
    int i = blockIdx.x * 256 + threadIdx.x;
    if (i >= C) return;
    float s = g[i] * rsqrtf(v[i] + 1e-5f);
    sc[i] = s; sh[i] = b[i] - m[i] * s;
}

// X (3200,5,5,6,6) NCDHW f32 -> NDHWC f16 (3200,5,6,6,5)
__global__ __launch_bounds__(256) void k_stage_input(const float* __restrict__ X,
                                                     half_t* __restrict__ xh) {
    long i = (long)blockIdx.x * 256 + threadIdx.x;
    if (i >= (long)NS * 900) return;
    int ci = (int)(i % 5);
    int w  = (int)((i / 5) % 6);
    int h  = (int)((i / 30) % 6);
    int d  = (int)((i / 180) % 5);
    long n = i / 900;
    xh[i] = (half_t)X[(((n * 5 + ci) * 5 + d) * 6 + h) * 6 + w];
}

// im2col (3x3x3, pad 1) from NDHWC f16 activations; K = Cin*27, zero-padded Kp
__global__ __launch_bounds__(256) void k_im2col(const half_t* __restrict__ act,
                                                half_t* __restrict__ A,
                                                int n0, int nCnt,
                                                int D, int H, int W, int Cin,
                                                int K, int Kp) {
    long i = (long)blockIdx.x * 256 + threadIdx.x;
    long tot = (long)nCnt * D * H * W * Kp;
    if (i >= tot) return;
    int  k = (int)(i % Kp);
    long m = i / Kp;
    int DHW = D * H * W;
    int pix = (int)(m % DHW);
    int ln  = (int)(m / DHW);
    half_t val = (half_t)0.f;
    if (k < K) {
        int ci = k / 27, rr = k % 27;
        int kd = rr / 9, kh = (rr / 3) % 3, kw = rr % 3;
        int ow = pix % W, oh = (pix / W) % H, od = pix / (W * H);
        int id = od + kd - 1, ih = oh + kh - 1, iw = ow + kw - 1;
        if (id >= 0 && id < D && ih >= 0 && ih < H && iw >= 0 && iw < W) {
            long n = n0 + ln;
            val = act[((((size_t)n * D + id) * H + ih) * W + iw) * Cin + ci];
        }
    }
    A[(size_t)m * Kp + k] = val;
}

// maxpool 2x2x2 stride 2 pad (1,1,1): (3200,5,6,6,64)->(3200,3,4,4,64) NDHWC
__global__ __launch_bounds__(256) void k_pool1(const half_t* __restrict__ in,
                                               half_t* __restrict__ out) {
    long i = (long)blockIdx.x * 256 + threadIdx.x;
    if (i >= (long)NS * 3 * 4 * 4 * 64) return;
    int c  = (int)(i % 64);
    int ow = (int)((i / 64) % 4);
    int oh = (int)((i / 256) % 4);
    int od = (int)((i / 1024) % 3);
    long n = i / 3072;
    float mx = -3.4e38f;
    for (int dd = 0; dd < 2; ++dd) {
        int d = od * 2 - 1 + dd; if (d < 0 || d >= 5) continue;
        for (int hh = 0; hh < 2; ++hh) {
            int h = oh * 2 - 1 + hh; if (h < 0 || h >= 6) continue;
            for (int ww = 0; ww < 2; ++ww) {
                int w = ow * 2 - 1 + ww; if (w < 0 || w >= 6) continue;
                float v = (float)in[(((n * 5 + d) * 6 + h) * 6 + w) * 64 + c];
                mx = fmaxf(mx, v);
            }
        }
    }
    out[i] = (half_t)mx;
}

// maxpool pad (1,0,0): in NDHWC (3200,3,4,4,256) -> out NCDHW flat (3200,2048)
__global__ __launch_bounds__(256) void k_pool2(const half_t* __restrict__ in,
                                               half_t* __restrict__ out) {
    long i = (long)blockIdx.x * 256 + threadIdx.x;
    if (i >= (long)NS * 2048) return;
    int ow = (int)(i % 2);
    int oh = (int)((i / 2) % 2);
    int od = (int)((i / 4) % 2);
    int c  = (int)((i / 8) % 256);
    long n = i / 2048;
    float mx = -3.4e38f;
    for (int dd = 0; dd < 2; ++dd) {
        int d = od * 2 - 1 + dd; if (d < 0 || d >= 3) continue;
        for (int hh = 0; hh < 2; ++hh) {
            int h = oh * 2 + hh; if (h >= 4) continue;
            for (int ww = 0; ww < 2; ++ww) {
                int w = ow * 2 + ww; if (w >= 4) continue;
                float v = (float)in[(((n * 3 + d) * 4 + h) * 4 + w) * 256 + c];
                mx = fmaxf(mx, v);
            }
        }
    }
    out[(size_t)n * 2048 + ((c * 2 + od) * 2 + oh) * 2 + ow] = (half_t)mx;
}

// GRU pointwise step: consumes gi row (bg*25+t), gh row bg; updates h, writes hs
__global__ __launch_bounds__(256) void k_gru_point(const float* __restrict__ gi,
                                                   const float* __restrict__ gh,
                                                   half_t* __restrict__ h,
                                                   half_t* __restrict__ hs, int t) {
    int i = blockIdx.x * 256 + threadIdx.x;
    if (i >= 128 * 256) return;
    int bg = i >> 8, c = i & 255;
    long row = (long)bg * 25 + t;
    const float* gir = gi + row * 768;
    const float* ghr = gh + (long)bg * 768;
    float r = 1.f / (1.f + __expf(-(gir[c]       + ghr[c])));
    float z = 1.f / (1.f + __expf(-(gir[256 + c] + ghr[256 + c])));
    float n = tanhf(gir[512 + c] + r * ghr[512 + c]);
    float hn = (1.f - z) * n + z * (float)h[i];
    h[i] = (half_t)hn;
    hs[row * 256 + c] = (half_t)hn;
}

// feats = bng0([node_x, enc[cell_idx]]) -> f16, K padded 513 -> 544
__global__ __launch_bounds__(256) void k_feats(const float* __restrict__ node_x,
                                               const float* __restrict__ enc,
                                               const int* __restrict__ cell,
                                               const float* __restrict__ sc,
                                               const float* __restrict__ sh,
                                               half_t* __restrict__ feats) {
    long i = (long)blockIdx.x * 256 + threadIdx.x;
    if (i >= (long)NN * 544) return;
    int c = (int)(i % 544);
    long n = i / 544;
    float v = 0.f;
    if (c == 0)            v = node_x[n] * sc[0] + sh[0];
    else if (c <= 512)     v = enc[(size_t)cell[n] * 512 + (c - 1)] * sc[c] + sh[c];
    feats[i] = (half_t)v;
}

// degree
__global__ __launch_bounds__(256) void k_deg(const int* __restrict__ dst, float* deg) {
    long i = (long)blockIdx.x * 256 + threadIdx.x;
    if (i >= NE) return;
    atomicAdd(&deg[dst[i]], 1.0f);
}

// GAT edge logits (ch=128): one wave per (edge, head)
__global__ __launch_bounds__(256) void k_edge_logits(
    const float* __restrict__ xl, const float* __restrict__ xr,
    const float* __restrict__ ea, const float* __restrict__ we,
    const float* __restrict__ att, const int* __restrict__ src,
    const int* __restrict__ dst, float* __restrict__ logit,
    unsigned* __restrict__ amax, int heads) {
    int unit = blockIdx.x * 8 + (threadIdx.x >> 5);
    int lane = threadIdx.x & 31;
    if (unit >= NE * heads) return;
    int ed = unit / heads, h = unit % heads;
    int s = src[ed], d = dst[ed];
    int NH = heads * 128;
    float a0 = ea[2 * ed], a1 = ea[2 * ed + 1];
    float acc = 0.f;
#pragma unroll
    for (int j = 0; j < 4; ++j) {
        int c = h * 128 + lane + j * 32;
        float e  = a0 * we[2 * c] + a1 * we[2 * c + 1];
        float sv = xl[(size_t)s * NH + c] + xr[(size_t)d * NH + c] + e;
        float g  = sv > 0.f ? sv : 0.2f * sv;
        acc += g * att[c];
    }
    for (int m = 16; m > 0; m >>= 1) acc += __shfl_xor(acc, m, 32);
    if (lane == 0) {
        logit[(size_t)ed * heads + h] = acc;
        atomicMax(&amax[(size_t)d * heads + h], f2ord(acc));
    }
}

// scalar (ch=1) edge logits for layer 3
__global__ __launch_bounds__(256) void k_edge_logits1(
    const float* __restrict__ xl, const float* __restrict__ xr,
    const float* __restrict__ ea, const float* __restrict__ we,
    const float* __restrict__ att, const int* __restrict__ src,
    const int* __restrict__ dst, float* __restrict__ logit,
    unsigned* __restrict__ amax) {
    long i = (long)blockIdx.x * 256 + threadIdx.x;
    if (i >= NE) return;
    float e  = ea[2 * i] * we[0] + ea[2 * i + 1] * we[1];
    float sv = xl[src[i]] + xr[dst[i]] + e;
    float g  = sv > 0.f ? sv : 0.2f * sv;
    float lg = g * att[0];
    logit[i] = lg;
    atomicMax(&amax[dst[i]], f2ord(lg));
}

// ex = exp(logit - amax[dst]); den[dst] += ex
__global__ __launch_bounds__(256) void k_edge_den(const float* __restrict__ logit,
                                                  const unsigned* __restrict__ amax,
                                                  const int* __restrict__ dst,
                                                  float* __restrict__ ex,
                                                  float* __restrict__ den, int heads) {
    long i = (long)blockIdx.x * 256 + threadIdx.x;
    if (i >= (long)NE * heads) return;
    long ed = i / heads; int h = (int)(i % heads);
    int d = dst[ed];
    float am = ord2f(amax[(size_t)d * heads + h]);
    float e  = __expf(logit[i] - am);
    ex[i] = e;
    atomicAdd(&den[(size_t)d * heads + h], e);
}

// out[dst, c] += xl[src, c] * alpha  (thread per edge-channel)
__global__ __launch_bounds__(256) void k_edge_agg(const float* __restrict__ xl,
                                                  const float* __restrict__ ex,
                                                  const float* __restrict__ den,
                                                  const int* __restrict__ src,
                                                  const int* __restrict__ dst,
                                                  float* __restrict__ out,
                                                  int NH, int ch, int heads) {
    long i = (long)blockIdx.x * 256 + threadIdx.x;
    if (i >= (long)NE * NH) return;
    long ed = i / NH; int c = (int)(i % NH);
    int h = c / ch;
    int s = src[ed], d = dst[ed];
    float alpha = ex[(size_t)ed * heads + h] / (den[(size_t)d * heads + h] + 1e-16f);
    atomicAdd(&out[(size_t)d * NH + c], xl[(size_t)s * NH + c] * alpha);
}

// out = bn(out/deg + bias) [relu] -> f16
__global__ __launch_bounds__(256) void k_node_finish(const float* __restrict__ out,
                                                     const float* __restrict__ deg,
                                                     const float* __restrict__ gb,
                                                     const float* __restrict__ sc,
                                                     const float* __restrict__ sh,
                                                     half_t* __restrict__ xg, int NH) {
    long i = (long)blockIdx.x * 256 + threadIdx.x;
    if (i >= (long)NN * NH) return;
    int c = (int)(i % NH);
    long n = i / NH;
    float v = out[i] / fmaxf(deg[n], 1.f) + gb[c];
    v = v * sc[c] + sh[c];
    v = v > 0.f ? v : 0.f;
    xg[i] = (half_t)v;
}

// layer-3 projections: xl3/xr3 = xg2 @ wl^T + bl  (K=128, N=1)
__global__ __launch_bounds__(256) void k_proj3(const half_t* __restrict__ xg,
                                               const float* __restrict__ wl,
                                               const float* __restrict__ wr,
                                               const float* __restrict__ bl,
                                               const float* __restrict__ br,
                                               float* __restrict__ xl3,
                                               float* __restrict__ xr3) {
    long i = (long)blockIdx.x * 256 + threadIdx.x;
    if (i >= NN) return;
    float al = bl[0], ar = br[0];
    const half_t* row = xg + (size_t)i * 128;
#pragma unroll 4
    for (int c = 0; c < 128; ++c) {
        float x = (float)row[c];
        al += x * wl[c];
        ar += x * wr[c];
    }
    xl3[i] = al; xr3[i] = ar;
}

// final: y = out3/deg + bias; masked
__global__ __launch_bounds__(256) void k_final(const float* __restrict__ out3,
                                               const float* __restrict__ deg,
                                               const float* __restrict__ bias,
                                               const unsigned char* __restrict__ mask,
                                               float* __restrict__ y) {
    long i = (long)blockIdx.x * 256 + threadIdx.x;
    if (i >= NN) return;
    float v = out3[i] / fmaxf(deg[i], 1.f) + bias[0];
    y[i] = mask[i] ? v : 0.f;
}

// ---------------------------------------------------------------------------
// Host side
// ---------------------------------------------------------------------------
static inline unsigned gblocks(long n) { return (unsigned)((n + 255) / 256); }

template <typename T>
static inline T* wsa(char*& p, size_t n) {
    T* r = (T*)p;
    size_t b = n * sizeof(T);
    p += (b + 255) & ~(size_t)255;
    return r;
}

static inline void gemm(hipStream_t st, const half_t* A, const half_t* B,
                        int M, int N, int Kp, const float* bias,
                        const float* sc, const float* sh, int relu,
                        float* oF, half_t* oH) {
    dim3 g(N / 64, (unsigned)((M + 127) / 128));
    k_gemm_nt<<<g, 256, 0, st>>>(A, B, M, N, Kp, bias, sc, sh, relu, oF, oH);
}

extern "C" void kernel_launch(void* const* d_in, const int* in_sizes, int n_in,
                              void* d_out, int out_size, void* d_ws, size_t ws_size,
                              hipStream_t stream) {
    (void)in_sizes; (void)n_in; (void)out_size; (void)ws_size;
    // ---- inputs ----
    const float* X        = (const float*)d_in[0];
    const float* node_x   = (const float*)d_in[1];
    const float* ea       = (const float*)d_in[2];
    const int*   cell_idx = (const int*)d_in[3];
    const int*   src      = (const int*)d_in[4];
    const int*   dstn     = src + NE;
    const unsigned char* mask = (const unsigned char*)d_in[5];
    float* y_out = (float*)d_out;

    // ---- params (dict insertion order) ----
    #define P(i) ((const float*)d_in[(i)])
    const float *conv1_w = P(6),  *conv1_b = P(7);
    const float *bn1g = P(8),  *bn1b = P(9),  *bn1m = P(10), *bn1v = P(11);
    const float *conv2_w = P(12), *conv2_b = P(13);
    const float *bn2g = P(14), *bn2b = P(15), *bn2m = P(16), *bn2v = P(17);
    const float *conv3_w = P(18), *conv3_b = P(19);
    const float *bn3g = P(20), *bn3b = P(21), *bn3m = P(22), *bn3v = P(23);
    const float *fc1_w = P(24), *fc1_b = P(25);
    const float *bn4g = P(26), *bn4b = P(27), *bn4m = P(28), *bn4v = P(29);
    const float *fc2_w = P(30), *fc2_b = P(31);
    const float *bn5g = P(32), *bn5b = P(33), *bn5m = P(34), *bn5v = P(35);
    const float *wih0 = P(36), *whh0 = P(37), *bih0 = P(38), *bhh0 = P(39);
    const float *wih1 = P(40), *whh1 = P(41), *bih1 = P(42), *bhh1 = P(43);
    const float *dense_w = P(44), *dense_b = P(45);
    const float *bg0g = P(46), *bg0b = P(47), *bg0m = P(48), *bg0v = P(49);
    const float *g1_wl = P(50), *g1_wr = P(51), *g1_bl = P(52), *g1_br = P(53);
    const float *g1_we = P(54), *g1_att = P(55), *g1_bias = P(56);
    const float *bg1g = P(57), *bg1b = P(58), *bg1m = P(59), *bg1v = P(60);
    const float *g2_wl = P(61), *g2_wr = P(62), *g2_bl = P(63), *g2_br = P(64);
    const float *g2_we = P(65), *g2_att = P(66), *g2_bias = P(67);
    const float *bg2g = P(68), *bg2b = P(69), *bg2m = P(70), *bg2v = P(71);
    const float *g3_wl = P(72), *g3_wr = P(73), *g3_bl = P(74), *g3_br = P(75);
    const float *g3_we = P(76), *g3_att = P(77), *g3_bias = P(78);
    #undef P

    // ---- workspace (deterministic bump allocation) ----
    char* wp = (char*)d_ws;
    half_t* xh    = wsa<half_t>(wp, (size_t)NS * 900);          // staged input NDHWC
    half_t* w1h   = wsa<half_t>(wp, 64  * 160);
    half_t* w2h   = wsa<half_t>(wp, 64  * 1728);
    half_t* w3h   = wsa<half_t>(wp, 256 * 1728);
    half_t* fc1h  = wsa<half_t>(wp, 512 * 2048);
    half_t* fc2h  = wsa<half_t>(wp, 256 * 512);
    half_t* wih0h = wsa<half_t>(wp, 768 * 256);
    half_t* whh0h = wsa<half_t>(wp, 768 * 256);
    half_t* wih1h = wsa<half_t>(wp, 768 * 256);
    half_t* whh1h = wsa<half_t>(wp, 768 * 256);
    half_t* denseh= wsa<half_t>(wp, 512 * 6400);
    half_t* g1wlh = wsa<half_t>(wp, 256 * 544);
    half_t* g1wrh = wsa<half_t>(wp, 256 * 544);
    half_t* g2wlh = wsa<half_t>(wp, 128 * 256);
    half_t* g2wrh = wsa<half_t>(wp, 128 * 256);
    float* bnS[8], *bnH[8];
    int bnC[8] = {64, 64, 256, 512, 256, 513, 256, 128};
    for (int i = 0; i < 8; ++i) { bnS[i] = wsa<float>(wp, bnC[i]); bnH[i] = wsa<float>(wp, bnC[i]); }
    half_t* imbuf = wsa<half_t>(wp, (size_t)11520 * 1728);       // im2col chunk
    half_t* a1h   = wsa<half_t>(wp, (size_t)NS * 180 * 64);
    half_t* a2h   = wsa<half_t>(wp, (size_t)NS * 180 * 64);
    half_t* p1h   = wsa<half_t>(wp, (size_t)NS * 48 * 64);
    half_t* a3h   = wsa<half_t>(wp, (size_t)NS * 48 * 256);
    half_t* p2h   = wsa<half_t>(wp, (size_t)NS * 2048);
    half_t* fc1o  = wsa<half_t>(wp, (size_t)NS * 512);
    half_t* fc2o  = wsa<half_t>(wp, (size_t)NS * 256);
    float*  gi    = wsa<float>(wp, (size_t)NS * 768);
    float*  gh    = wsa<float>(wp, (size_t)128 * 768);
    half_t* hprev = wsa<half_t>(wp, 128 * 256);
    half_t* hs0   = wsa<half_t>(wp, (size_t)NS * 256);
    half_t* hs1   = wsa<half_t>(wp, (size_t)NS * 256);
    float*  enc   = wsa<float>(wp, (size_t)128 * 512);
    half_t* featsh= wsa<half_t>(wp, (size_t)NN * 544);
    float*  xl1   = wsa<float>(wp, (size_t)NN * 256);
    float*  xr1   = wsa<float>(wp, (size_t)NN * 256);
    float*  out1  = wsa<float>(wp, (size_t)NN * 256);
    half_t* xg1h  = wsa<half_t>(wp, (size_t)NN * 256);
    float*  xl2   = wsa<float>(wp, (size_t)NN * 128);
    float*  xr2   = wsa<float>(wp, (size_t)NN * 128);
    float*  out2  = wsa<float>(wp, (size_t)NN * 128);
    half_t* xg2h  = wsa<half_t>(wp, (size_t)NN * 128);
    float*  xl3   = wsa<float>(wp, NN);
    float*  xr3   = wsa<float>(wp, NN);
    float*  out3  = wsa<float>(wp, NN);
    float*  logit = wsa<float>(wp, (size_t)NE * 2);
    float*  exb   = wsa<float>(wp, (size_t)NE * 2);
    unsigned* amax= wsa<unsigned>(wp, (size_t)NN * 2);
    float*  den   = wsa<float>(wp, (size_t)NN * 2);
    float*  deg   = wsa<float>(wp, NN);

    #define FILL0(ptr, words) k_fill_u32<<<gblocks(words), 256, 0, stream>>>((unsigned*)(ptr), 0u, (long)(words))

    // ---- BN folds ----
    k_bnprep<<<1, 64,  0, stream>>>(bn1g, bn1b, bn1m, bn1v, bnS[0], bnH[0], 64);
    k_bnprep<<<1, 64,  0, stream>>>(bn2g, bn2b, bn2m, bn2v, bnS[1], bnH[1], 64);
    k_bnprep<<<1, 256, 0, stream>>>(bn3g, bn3b, bn3m, bn3v, bnS[2], bnH[2], 256);
    k_bnprep<<<2, 256, 0, stream>>>(bn4g, bn4b, bn4m, bn4v, bnS[3], bnH[3], 512);
    k_bnprep<<<1, 256, 0, stream>>>(bn5g, bn5b, bn5m, bn5v, bnS[4], bnH[4], 256);
    k_bnprep<<<3, 256, 0, stream>>>(bg0g, bg0b, bg0m, bg0v, bnS[5], bnH[5], 513);
    k_bnprep<<<1, 256, 0, stream>>>(bg1g, bg1b, bg1m, bg1v, bnS[6], bnH[6], 256);
    k_bnprep<<<1, 128, 0, stream>>>(bg2g, bg2b, bg2m, bg2v, bnS[7], bnH[7], 128);

    // ---- weight conversions (f32 -> f16, K padded to /32) ----
    #define CVT(s, d, r, K, Kp) k_cvt_pad<<<gblocks((long)(r)*(Kp)), 256, 0, stream>>>((s), (d), (r), (K), (Kp))
    CVT(conv1_w, w1h, 64, 135, 160);
    CVT(conv2_w, w2h, 64, 1728, 1728);
    CVT(conv3_w, w3h, 256, 1728, 1728);
    CVT(fc1_w, fc1h, 512, 2048, 2048);
    CVT(fc2_w, fc2h, 256, 512, 512);
    CVT(wih0, wih0h, 768, 256, 256);  CVT(whh0, whh0h, 768, 256, 256);
    CVT(wih1, wih1h, 768, 256, 256);  CVT(whh1, whh1h, 768, 256, 256);
    CVT(dense_w, denseh, 512, 6400, 6400);
    CVT(g1_wl, g1wlh, 256, 513, 544); CVT(g1_wr, g1wrh, 256, 513, 544);
    CVT(g2_wl, g2wlh, 128, 256, 256); CVT(g2_wr, g2wrh, 128, 256, 256);
    #undef CVT

    // ---- stage input ----
    k_stage_input<<<gblocks((long)NS * 900), 256, 0, stream>>>(X, xh);

    // ---- conv1 / conv2 / conv3 via im2col + WMMA GEMM (chunks of 64 samples) ----
    for (int n0 = 0; n0 < NS; n0 += 64) {
        k_im2col<<<gblocks((long)64 * 180 * 160), 256, 0, stream>>>(xh, imbuf, n0, 64, 5, 6, 6, 5, 135, 160);
        gemm(stream, imbuf, w1h, 64 * 180, 64, 160, conv1_b, bnS[0], bnH[0], 1,
             nullptr, a1h + (size_t)n0 * 180 * 64);
    }
    for (int n0 = 0; n0 < NS; n0 += 64) {
        k_im2col<<<gblocks((long)64 * 180 * 1728), 256, 0, stream>>>(a1h, imbuf, n0, 64, 5, 6, 6, 64, 1728, 1728);
        gemm(stream, imbuf, w2h, 64 * 180, 64, 1728, conv2_b, bnS[1], bnH[1], 1,
             nullptr, a2h + (size_t)n0 * 180 * 64);
    }
    k_pool1<<<gblocks((long)NS * 3072), 256, 0, stream>>>(a2h, p1h);
    for (int n0 = 0; n0 < NS; n0 += 64) {
        k_im2col<<<gblocks((long)64 * 48 * 1728), 256, 0, stream>>>(p1h, imbuf, n0, 64, 3, 4, 4, 64, 1728, 1728);
        gemm(stream, imbuf, w3h, 64 * 48, 256, 1728, conv3_b, bnS[2], bnH[2], 1,
             nullptr, a3h + (size_t)n0 * 48 * 256);
    }
    k_pool2<<<gblocks((long)NS * 2048), 256, 0, stream>>>(a3h, p2h);

    // ---- fc1 / fc2 ----
    gemm(stream, p2h,  fc1h, NS, 512, 2048, fc1_b, bnS[3], bnH[3], 1, nullptr, fc1o);
    gemm(stream, fc1o, fc2h, NS, 256, 512,  fc2_b, bnS[4], bnH[4], 1, nullptr, fc2o);

    // ---- GRU (2 layers, T=25) ----
    const half_t* xin[2]   = {fc2o, hs0};
    half_t*       hsout[2] = {hs0, hs1};
    const half_t* wihh[2]  = {wih0h, wih1h};
    const half_t* whhh[2]  = {whh0h, whh1h};
    const float*  bih[2]   = {bih0, bih1};
    const float*  bhh[2]   = {bhh0, bhh1};
    for (int l = 0; l < 2; ++l) {
        gemm(stream, xin[l], wihh[l], NS, 768, 256, bih[l], nullptr, nullptr, 0, gi, nullptr);
        FILL0(hprev, 128 * 256 / 2);
        for (int t = 0; t < 25; ++t) {
            gemm(stream, hprev, whhh[l], 128, 768, 256, bhh[l], nullptr, nullptr, 0, gh, nullptr);
            k_gru_point<<<gblocks(128 * 256), 256, 0, stream>>>(gi, gh, hprev, hsout[l], t);
        }
    }

    // ---- dense encoder ----
    gemm(stream, hs1, denseh, 128, 512, 6400, dense_b, nullptr, nullptr, 1, enc, nullptr);

    // ---- features + degree ----
    k_feats<<<gblocks((long)NN * 544), 256, 0, stream>>>(node_x, enc, cell_idx, bnS[5], bnH[5], featsh);
    FILL0(deg, NN);
    k_deg<<<gblocks(NE), 256, 0, stream>>>(dstn, deg);

    // ---- GAT layer 1 (heads=2, ch=128) ----
    gemm(stream, featsh, g1wlh, NN, 256, 544, g1_bl, nullptr, nullptr, 0, xl1, nullptr);
    gemm(stream, featsh, g1wrh, NN, 256, 544, g1_br, nullptr, nullptr, 0, xr1, nullptr);
    FILL0(amax, (long)NN * 2); FILL0(den, (long)NN * 2); FILL0(out1, (long)NN * 256);
    k_edge_logits<<<gblocks((long)NE * 2 * 32) , 256, 0, stream>>>(xl1, xr1, ea, g1_we, g1_att,
                                                                   src, dstn, logit, amax, 2);
    k_edge_den<<<gblocks((long)NE * 2), 256, 0, stream>>>(logit, amax, dstn, exb, den, 2);
    k_edge_agg<<<gblocks((long)NE * 256), 256, 0, stream>>>(xl1, exb, den, src, dstn, out1, 256, 128, 2);
    k_node_finish<<<gblocks((long)NN * 256), 256, 0, stream>>>(out1, deg, g1_bias, bnS[6], bnH[6], xg1h, 256);

    // ---- GAT layer 2 (heads=1, ch=128) ----
    gemm(stream, xg1h, g2wlh, NN, 128, 256, g2_bl, nullptr, nullptr, 0, xl2, nullptr);
    gemm(stream, xg1h, g2wrh, NN, 128, 256, g2_br, nullptr, nullptr, 0, xr2, nullptr);
    FILL0(amax, NN); FILL0(den, NN); FILL0(out2, (long)NN * 128);
    k_edge_logits<<<gblocks((long)NE * 32), 256, 0, stream>>>(xl2, xr2, ea, g2_we, g2_att,
                                                              src, dstn, logit, amax, 1);
    k_edge_den<<<gblocks(NE), 256, 0, stream>>>(logit, amax, dstn, exb, den, 1);
    k_edge_agg<<<gblocks((long)NE * 128), 256, 0, stream>>>(xl2, exb, den, src, dstn, out2, 128, 128, 1);
    k_node_finish<<<gblocks((long)NN * 128), 256, 0, stream>>>(out2, deg, g2_bias, bnS[7], bnH[7], xg2h, 128);

    // ---- GAT layer 3 (heads=1, ch=1) ----
    k_proj3<<<gblocks(NN), 256, 0, stream>>>(xg2h, g3_wl, g3_wr, g3_bl, g3_br, xl3, xr3);
    FILL0(amax, NN); FILL0(den, NN); FILL0(out3, NN);
    k_edge_logits1<<<gblocks(NE), 256, 0, stream>>>(xl3, xr3, ea, g3_we, g3_att, src, dstn, logit, amax);
    k_edge_den<<<gblocks(NE), 256, 0, stream>>>(logit, amax, dstn, exb, den, 1);
    k_edge_agg<<<gblocks(NE), 256, 0, stream>>>(xl3, exb, den, src, dstn, out3, 1, 1, 1);
    k_final<<<gblocks(NN), 256, 0, stream>>>(out3, deg, g3_bias, mask, y_out);

    #undef FILL0
}
